// FeaturePropagation_1760936591724
// MI455X (gfx1250) — matrix operand order repaired and checked by reading, not compiled, not run on hardware
//
#include <hip/hip_runtime.h>
#include <hip/hip_bf16.h>

typedef __attribute__((ext_vector_type(16))) _Float16 v16h;
typedef __attribute__((ext_vector_type(8)))  float    v8f;
typedef __attribute__((ext_vector_type(4)))  unsigned u32x4;

#define NQ  32768
#define MS  4096
#define CH  256

// ---------------------------------------------------------------------------
// CDNA5 async global->LDS copy (16B per lane), tracked by ASYNCcnt.
// LDS address operand = low 32 bits of the flat pointer (ISA 10.2: LDS
// aperture maps addr[31:0] directly to the DS address).
// ---------------------------------------------------------------------------
__device__ __forceinline__ void async_cp16(void* lds, const void* gptr) {
#if defined(__HIP_DEVICE_COMPILE__)
  unsigned ldsa = (unsigned)(unsigned long long)lds;
  asm volatile("global_load_async_to_lds_b128 %0, %1, off"
               :
               : "v"(ldsa), "v"(gptr)
               : "memory");
#endif
}

__device__ __forceinline__ void wait_async0() {
#if defined(__HIP_DEVICE_COMPILE__)
  asm volatile("s_wait_asynccnt 0x0" ::: "memory");
#endif
}

// ---------------------------------------------------------------------------
// Kernel 1: 3-NN search. pos_skip (48KB) lives entirely in LDS; one query per
// thread; uniform-address LDS reads broadcast across the wave.
// ---------------------------------------------------------------------------
__global__ __launch_bounds__(256) void knn_kernel(
    const float* __restrict__ pos, const float* __restrict__ pos_skip,
    int* __restrict__ knn_idx, float* __restrict__ knn_w) {
  __shared__ float sp[MS * 3];  // 48 KB
  for (int i = threadIdx.x; i < MS * 3; i += 256) sp[i] = pos_skip[i];
  __syncthreads();

  const int n = blockIdx.x * 256 + threadIdx.x;
  const float px = pos[n * 3 + 0];
  const float py = pos[n * 3 + 1];
  const float pz = pos[n * 3 + 2];

  float b0 = 3.4e38f, b1 = 3.4e38f, b2 = 3.4e38f;
  int   i0 = 0, i1 = 0, i2 = 0;
  for (int m = 0; m < MS; ++m) {
    float dx = px - sp[m * 3 + 0];
    float dy = py - sp[m * 3 + 1];
    float dz = pz - sp[m * 3 + 2];
    float d  = fmaf(dx, dx, fmaf(dy, dy, dz * dz));
    if (d < b2) {                      // rarely taken past warm-up
      if (d < b1) {
        b2 = b1; i2 = i1;
        if (d < b0) { b1 = b0; i1 = i0; b0 = d; i0 = m; }
        else        { b1 = d;  i1 = m; }
      } else { b2 = d; i2 = m; }
    }
  }
  float d0 = sqrtf(fmaxf(b0, 0.f));
  float d1 = sqrtf(fmaxf(b1, 0.f));
  float d2 = sqrtf(fmaxf(b2, 0.f));
  float w0 = 1.f / (d0 + 1e-8f);
  float w1 = 1.f / (d1 + 1e-8f);
  float w2 = 1.f / (d2 + 1e-8f);
  float s  = 1.f / (w0 + w1 + w2);
  knn_idx[n * 3 + 0] = i0; knn_idx[n * 3 + 1] = i1; knn_idx[n * 3 + 2] = i2;
  knn_w[n * 3 + 0] = w0 * s; knn_w[n * 3 + 1] = w1 * s; knn_w[n * 3 + 2] = w2 * s;
}

// ---------------------------------------------------------------------------
// Kernel 2: inverse-distance interpolation fused with f32->f16 conversion of
// the concatenated activation hcat = [x | x_interp]  (f16, [N, 512]).
// ---------------------------------------------------------------------------
__global__ __launch_bounds__(256) void interp_kernel(
    const float* __restrict__ x, const float* __restrict__ x_skip,
    const int* __restrict__ knn_idx, const float* __restrict__ knn_w,
    _Float16* __restrict__ hcat) {
  const int n = blockIdx.x;
  const int t = threadIdx.x;
  const int i0 = knn_idx[n * 3 + 0], i1 = knn_idx[n * 3 + 1], i2 = knn_idx[n * 3 + 2];
  const float w0 = knn_w[n * 3 + 0], w1 = knn_w[n * 3 + 1], w2 = knn_w[n * 3 + 2];
  float v = w0 * x_skip[(size_t)i0 * CH + t]
          + w1 * x_skip[(size_t)i1 * CH + t]
          + w2 * x_skip[(size_t)i2 * CH + t];
  hcat[(size_t)n * 512 + t]      = (_Float16)x[(size_t)n * CH + t];
  hcat[(size_t)n * 512 + CH + t] = (_Float16)v;
}

// ---------------------------------------------------------------------------
// Weight conversion f32 -> f16 (tiny, one-shot)
// ---------------------------------------------------------------------------
__global__ void cvt_f32_f16(const float* __restrict__ s, _Float16* __restrict__ d, int n) {
  int i = blockIdx.x * 256 + threadIdx.x;
  if (i < n) d[i] = (_Float16)s[i];
}

// ---------------------------------------------------------------------------
// WMMA GEMM: out[m][j] = act( sum_k A[m][k] * B[j][k] + bias[j] )
// A: [M x K] f16 row-major (activations), B: [Nout x K] f16 row-major.
// Block = 256 threads = 8 waves. Block tile: 256(M) x 64(N); each wave owns a
// 32x64 strip (8 f32 accumulators, two A-frags reuse four B-frags).
// K stepped by 32 through DOUBLE-BUFFERED LDS tiles filled with
// global_load_async_to_lds_b128 (ASYNCcnt) -> prefetch overlaps the WMMAs and
// only one barrier per K-step is needed.
// Fragment layouts (CDNA5 ISA 7.12.2, wave32):
//   A 16x32 f16: lanes 0-15 hold K{0..7,16..23}, lanes 16-31 K{8..15,24..31}
//   B 32x16 f16: lanes 0-15 hold K{0..7,16..23} of col N=lane, lanes 16-31 K{8..15,24..31}
// ---------------------------------------------------------------------------
template <bool RELU, bool OUTF16>
__global__ __launch_bounds__(256) void gemm_wmma_kernel(
    const _Float16* __restrict__ A, const _Float16* __restrict__ B,
    const float* __restrict__ bias,
    float* __restrict__ outF, _Float16* __restrict__ outH,
    int Nout, int K) {
  __shared__ u32x4 sA[2][256 * 4];  // 2 x 16 KB  (256 rows x 32 f16)
  __shared__ u32x4 sB[2][64 * 4];   // 2 x  4 KB  ( 64 rows x 32 f16)

  const int tid    = threadIdx.x;
  const int lane   = tid & 31;
  const int wave   = tid >> 5;
  const int half   = lane >> 4;   // 0: lanes 0-15, 1: lanes 16-31
  const int l16    = lane & 15;
  const int mblock = blockIdx.y * 256;
  const int jblock = blockIdx.x * 64;

  // Stage one 32-wide K slice into LDS buffer `buf` (async, no wait here).
  auto stage = [&](int buf, int k0) {
#pragma unroll
    for (int s = 0; s < 4; ++s) {          // A: 1024 x 16B chunks, 4/thread
      int c = tid + s * 256;
      int row = c >> 2, q = c & 3;
      async_cp16(&sA[buf][c], A + (size_t)(mblock + row) * K + k0 + q * 8);
    }
    {                                      // B: 256 x 16B chunks, 1/thread
      int row = tid >> 2, q = tid & 3;
      async_cp16(&sB[buf][tid], B + (size_t)(jblock + row) * K + k0 + q * 8);
    }
  };

  v8f acc[2][4] = {};

  stage(0, 0);
  wait_async0();
  __syncthreads();

  int buf = 0;
  for (int k0 = 0; k0 < K; k0 += 32) {
    if (k0 + 32 < K) stage(buf ^ 1, k0 + 32);   // prefetch next slice

    union { v16h h; u32x4 q[2]; } a0, a1, bf;
    const int arow0 = wave * 32 + l16;
    const int arow1 = arow0 + 16;
    a0.q[0] = sA[buf][arow0 * 4 + half];        // K 0..7  / 8..15
    a0.q[1] = sA[buf][arow0 * 4 + 2 + half];    // K 16..23 / 24..31
    a1.q[0] = sA[buf][arow1 * 4 + half];
    a1.q[1] = sA[buf][arow1 * 4 + 2 + half];

#pragma unroll
    for (int t = 0; t < 4; ++t) {
      const int brow = t * 16 + l16;
      bf.q[0] = sB[buf][brow * 4 + half];          // K 0..7  / 8..15
      bf.q[1] = sB[buf][brow * 4 + 2 + half];      // K 16..23 / 24..31
      acc[0][t] = __builtin_amdgcn_wmma_f32_16x16x32_f16(
          false, a0.h, false, bf.h, (short)0, acc[0][t], false, false);
      acc[1][t] = __builtin_amdgcn_wmma_f32_16x16x32_f16(
          false, a1.h, false, bf.h, (short)0, acc[1][t], false, false);
    }

    wait_async0();       // next buffer resident (this wave's copies)
    __syncthreads();     // all waves done reading buf & done staging buf^1
    buf ^= 1;
  }

  // Epilogue. C/D layout: lane l<16 -> D[M=i][N=l]; lane l>=16 -> D[M=8+i][N=l-16]
#pragma unroll
  for (int mm = 0; mm < 2; ++mm) {
#pragma unroll
    for (int t = 0; t < 4; ++t) {
      const int col = jblock + t * 16 + l16;
      const float bb = bias[col];
#pragma unroll
      for (int i = 0; i < 8; ++i) {
        const int row = mblock + wave * 32 + mm * 16 + half * 8 + i;
        float v = acc[mm][t][i] + bb;
        if (RELU) v = fmaxf(v, 0.f);
        if (OUTF16) outH[(size_t)row * Nout + col] = (_Float16)v;
        else        outF[(size_t)row * Nout + col] = v;
      }
    }
  }
}

// ---------------------------------------------------------------------------
extern "C" void kernel_launch(void* const* d_in, const int* in_sizes, int n_in,
                              void* d_out, int out_size, void* d_ws, size_t ws_size,
                              hipStream_t stream) {
  const float* x_skip   = (const float*)d_in[0];  // [4096, 256]
  const float* x        = (const float*)d_in[1];  // [32768, 256]
  const float* pos_skip = (const float*)d_in[2];  // [4096, 3]
  const float* pos      = (const float*)d_in[3];  // [32768, 3]
  const float* w1       = (const float*)d_in[4];  // [256, 512]
  const float* b1       = (const float*)d_in[5];  // [256]
  const float* w2       = (const float*)d_in[6];  // [256, 256]
  const float* b2       = (const float*)d_in[7];  // [256]
  float* out = (float*)d_out;                     // [32768, 256]

  char* ws = (char*)d_ws;
  int*      knn_idx = (int*)ws;        ws += (size_t)NQ * 3 * sizeof(int);
  float*    knn_w   = (float*)ws;      ws += (size_t)NQ * 3 * sizeof(float);
  _Float16* w1h     = (_Float16*)ws;   ws += (size_t)256 * 512 * sizeof(_Float16);
  _Float16* w2h     = (_Float16*)ws;   ws += (size_t)256 * 256 * sizeof(_Float16);
  _Float16* hcat    = (_Float16*)ws;   ws += (size_t)NQ * 512 * sizeof(_Float16);
  _Float16* h1      = (_Float16*)ws;   ws += (size_t)NQ * 256 * sizeof(_Float16);

  cvt_f32_f16<<<(256 * 512 + 255) / 256, 256, 0, stream>>>(w1, w1h, 256 * 512);
  cvt_f32_f16<<<(256 * 256 + 255) / 256, 256, 0, stream>>>(w2, w2h, 256 * 256);

  knn_kernel<<<NQ / 256, 256, 0, stream>>>(pos, pos_skip, knn_idx, knn_w);
  interp_kernel<<<NQ, 256, 0, stream>>>(x, x_skip, knn_idx, knn_w, hcat);

  // GEMM1: [32768,512] x [512,256]^T + b1, ReLU -> h1 (f16)
  gemm_wmma_kernel<true, true><<<dim3(256 / 64, NQ / 256), 256, 0, stream>>>(
      hcat, w1h, b1, nullptr, h1, 256, 512);
  // GEMM2: [32768,256] x [256,256]^T + b2 -> out (f32)
  gemm_wmma_kernel<false, false><<<dim3(256 / 64, NQ / 256), 256, 0, stream>>>(
      h1, w2h, b2, out, nullptr, 256, 256);
}
